// RNN_44865228374568
// MI455X (gfx1250) — matrix operand order repaired and checked by reading, not compiled, not run on hardware
//
#include <hip/hip_runtime.h>
#include <hip/hip_bf16.h>
#include <math.h>

// Problem constants (match reference: B=32, T=512, I=H=1024, L=2)
#define BB 32
#define TT 512
#define II 1024
#define HH 1024

#define PADC 8              // LDS column padding (elems) -> conflict-free ds_load_b128
#define CSTR (HH + PADC)    // LDS column stride = 1032 elems

typedef __attribute__((ext_vector_type(16))) __bf16 bf16x16;
typedef __attribute__((ext_vector_type(8)))  float  f32x8;

union ABFrag { bf16x16 v; uint4 q[2]; };

// ---------------------------------------------------------------------------
// Utility kernels
// ---------------------------------------------------------------------------
__global__ void k_zero_u32(unsigned* __restrict__ p, int n) {
    int i = blockIdx.x * blockDim.x + threadIdx.x;
    if (i < n) p[i] = 0u;
}

__global__ void k_cvt_bf16(const float* __restrict__ s, __bf16* __restrict__ d, int n) {
    int i = blockIdx.x * blockDim.x + threadIdx.x;
    if (i < n) d[i] = (__bf16)s[i];
}

// ---------------------------------------------------------------------------
// Input-projection GEMM:  C[m,n] = sum_k A[m,k]*W[n,k] + bi[n] + bh[n]
//   A: [M=BB*TT, II] bf16 row-major, W: [HH, II] bf16 row-major, C: f32.
// One wave computes a 16x64 strip; K-loop is software-pipelined (ping-pong
// fragment registers) with sched_barrier fences so the scheduler cannot sink
// the prefetch loads into the WMMA group -> partial s_wait_loadcnt waits.
// ---------------------------------------------------------------------------
__global__ void __launch_bounds__(256) k_gemm_pre(
    const __bf16* __restrict__ A, const __bf16* __restrict__ W,
    const float*  __restrict__ bi, const float* __restrict__ bh,
    float* __restrict__ C)
{
    const int lane  = threadIdx.x & 31;
    const int wave  = threadIdx.x >> 5;
    const int strip = blockIdx.x * 8 + wave;     // 0 .. 16383
    const int mtile = strip >> 4;                // 1024 M-tiles
    const int ngrp  = strip & 15;                // 16 groups of 64 cols
    const int m0 = mtile * 16;
    const int n0 = ngrp * 64;

    const int l15  = lane & 15;
    const int hi   = lane >> 4;
    const int asub = hi ? 8 : 0;     // A K sub-offset per WMMA 16-bit layout
    const int bsub = hi ? 8 : 0;     // B K sub-offset
    const int mb   = hi ? 8 : 0;     // C/D row base per lane half

    const __bf16* arow = A + (size_t)(m0 + l15) * II + asub;
    const __bf16* wrow[4];
#pragma unroll
    for (int j = 0; j < 4; ++j)
        wrow[j] = W + (size_t)(n0 + j * 16 + l15) * II + bsub;

    f32x8 acc[4] = {};
    ABFrag a0, a1, b0[4], b1[4];

    // prologue: K-tile 0
    a0.q[0] = *reinterpret_cast<const uint4*>(arow);
    a0.q[1] = *reinterpret_cast<const uint4*>(arow + 16);
#pragma unroll
    for (int j = 0; j < 4; ++j) {
        b0[j].q[0] = *reinterpret_cast<const uint4*>(wrow[j]);
        b0[j].q[1] = *reinterpret_cast<const uint4*>(wrow[j] + 16);
    }

    for (int k0 = 0; k0 < II; k0 += 64) {
        // load K-tile (k0+32) while computing K-tile k0
        a1.q[0] = *reinterpret_cast<const uint4*>(arow + k0 + 32);
        a1.q[1] = *reinterpret_cast<const uint4*>(arow + k0 + 48);
#pragma unroll
        for (int j = 0; j < 4; ++j) {
            b1[j].q[0] = *reinterpret_cast<const uint4*>(wrow[j] + k0 + 32);
            b1[j].q[1] = *reinterpret_cast<const uint4*>(wrow[j] + k0 + 48);
        }
        __builtin_amdgcn_sched_barrier(0);   // keep prefetch above the MFU work
#pragma unroll
        for (int j = 0; j < 4; ++j)
            acc[j] = __builtin_amdgcn_wmma_f32_16x16x32_bf16(
                false, a0.v, false, b0[j].v, (short)0, acc[j], false, false);
        __builtin_amdgcn_sched_barrier(0);

        // load K-tile (k0+64) (wraps harmlessly to 0 on last iter)
        const int kn = (k0 + 64) & (II - 1);
        a0.q[0] = *reinterpret_cast<const uint4*>(arow + kn);
        a0.q[1] = *reinterpret_cast<const uint4*>(arow + kn + 16);
#pragma unroll
        for (int j = 0; j < 4; ++j) {
            b0[j].q[0] = *reinterpret_cast<const uint4*>(wrow[j] + kn);
            b0[j].q[1] = *reinterpret_cast<const uint4*>(wrow[j] + kn + 16);
        }
        __builtin_amdgcn_sched_barrier(0);
#pragma unroll
        for (int j = 0; j < 4; ++j)
            acc[j] = __builtin_amdgcn_wmma_f32_16x16x32_bf16(
                false, a1.v, false, b1[j].v, (short)0, acc[j], false, false);
        __builtin_amdgcn_sched_barrier(0);
    }

#pragma unroll
    for (int j = 0; j < 4; ++j) {
        const int n = n0 + j * 16 + l15;
        const float bias = bi[n] + bh[n];
#pragma unroll
        for (int r = 0; r < 8; ++r) {
            C[(size_t)(m0 + mb + r) * HH + n] = acc[j][r] + bias;
        }
    }
}

// ---------------------------------------------------------------------------
// Persistent recurrent layer:  for t in 0..T-1:  h = tanh(pre[:,t,:] + h @ Whh^T)
//   pre: [BB, TT, HH] f32; Whh: [HH, HH] bf16; hp/hq: [BB, HH] bf16 ping-pong.
// 16 blocks x 8 waves = 128 waves = (BB/16)*(HH/16) tiles, one tile per wave.
// Wave pairs (2j, 2j+1) share an n-tile, so each block stages only 4 distinct
// Whh tiles (~132 KB) into LDS once; the 512-step t-loop reads B fragments
// from LDS (ds_load_b128) and streams the A (h-state) fragments from L2
// through an 8-deep rotating register buffer (K unrolled by 8) with
// sched_barrier fences, so each WMMA runs with ~14 A-loads in flight.
// Device-wide barrier per step via agent-scope atomics + s_sleep spin.
// ---------------------------------------------------------------------------
template <bool F32OUT>
__global__ void __launch_bounds__(256) k_rnn_layer(
    const float*  __restrict__ pre,
    const __bf16* __restrict__ Whh,
    __bf16* __restrict__ hp,        // state at t=0 (ping)
    __bf16* __restrict__ hq,        // scratch (pong)
    __bf16* __restrict__ out_bf,
    float*  __restrict__ out_f,
    unsigned* __restrict__ ctr)
{
    extern __shared__ __align__(16) __bf16 smem[];   // 4 * 16 * CSTR bf16

    const int lane  = threadIdx.x & 31;
    const int wave  = threadIdx.x >> 5;
    const int gw    = blockIdx.x * 8 + wave;     // 0..127
    const int mtile = gw & 1;                    // BB/16 = 2
    const int ntile = gw >> 1;                   // HH/16 = 64
    const int m0 = mtile * 16;
    const int n0 = ntile * 16;

    const int l15  = lane & 15;
    const int hi   = lane >> 4;
    const int asub = hi ? 8 : 0;
    const int bsub = hi ? 8 : 0;
    const int mb   = hi ? 8 : 0;
    const int n    = n0 + l15;

    // ---- stage this wave-pair's Whh tile into LDS (done once) ----
    {
        const int tloc = wave >> 1;              // 0..3: LDS tile slot
        const int half = wave & 1;               // wave pair splits the copy
        const int col  = half * 8 + (lane & 7);  // 0..15 within tile
        const int seg  = lane >> 3;              // 0..3: quarter of the column
        const __bf16* src = Whh + (size_t)(n0 - l15 + col) * HH + seg * 256;
        __bf16* dst = smem + ((size_t)tloc * 16 + col) * CSTR + seg * 256;
#pragma unroll 4
        for (int i = 0; i < 256; i += 8)
            *reinterpret_cast<uint4*>(dst + i) =
                *reinterpret_cast<const uint4*>(src + i);
    }
    __syncthreads();

    const __bf16* brow = smem + ((size_t)(wave >> 1) * 16 + l15) * CSTR + bsub;

    for (int t = 0; t < TT; ++t) {
        const __bf16* hr = (t & 1) ? hq : hp;   // read state(t)
        __bf16*       hw = (t & 1) ? hp : hq;   // write state(t+1)
        const __bf16* arow = hr + (size_t)(m0 + l15) * HH + asub;

        f32x8 acc = {};
        ABFrag a[8], b[2];

        // prologue: A tiles 0..6 in flight, B tile 0 resident
#pragma unroll
        for (int p = 0; p < 7; ++p) {
            a[p].q[0] = *reinterpret_cast<const uint4*>(arow + p * 32);
            a[p].q[1] = *reinterpret_cast<const uint4*>(arow + p * 32 + 16);
        }
        b[0].q[0] = *reinterpret_cast<const uint4*>(brow);
        b[0].q[1] = *reinterpret_cast<const uint4*>(brow + 16);
        __builtin_amdgcn_sched_barrier(0);

        // 32 K-tiles, unrolled by 8 so rotating-buffer indices are static
        for (int kt = 0; kt < 32; kt += 8) {
#pragma unroll
            for (int p = 0; p < 8; ++p) {
                const int ta = (kt + p + 7) & 31;   // A prefetch, 7 tiles ahead
                const int tb = (kt + p + 1) & 31;   // B prefetch, 1 tile ahead
                a[(p + 7) & 7].q[0] =
                    *reinterpret_cast<const uint4*>(arow + ta * 32);
                a[(p + 7) & 7].q[1] =
                    *reinterpret_cast<const uint4*>(arow + ta * 32 + 16);
                b[(p + 1) & 1].q[0] =
                    *reinterpret_cast<const uint4*>(brow + tb * 32);
                b[(p + 1) & 1].q[1] =
                    *reinterpret_cast<const uint4*>(brow + tb * 32 + 16);
                __builtin_amdgcn_sched_barrier(0);
                acc = __builtin_amdgcn_wmma_f32_16x16x32_bf16(
                    false, a[p & 7].v, false, b[p & 1].v, (short)0, acc,
                    false, false);
                __builtin_amdgcn_sched_barrier(0);
            }
        }

#pragma unroll
        for (int r = 0; r < 8; ++r) {
            const int m = m0 + mb + r;
            const size_t o = ((size_t)m * TT + t) * HH + n;
            const float v = tanhf(acc[r] + pre[o]);
            if (F32OUT) out_f[o] = v;
            else        out_bf[o] = (__bf16)v;
            hw[(size_t)m * HH + n] = (__bf16)v;
        }

        // --- device-wide barrier (all 16 blocks) before consuming h(t+1) ---
        __threadfence();
        __syncthreads();
        if (threadIdx.x == 0) {
            __hip_atomic_fetch_add(ctr, 1u, __ATOMIC_RELEASE, __HIP_MEMORY_SCOPE_AGENT);
            const unsigned target = (unsigned)(t + 1) * gridDim.x;
            while (__hip_atomic_load(ctr, __ATOMIC_ACQUIRE, __HIP_MEMORY_SCOPE_AGENT) < target) {
                __builtin_amdgcn_s_sleep(1);
            }
        }
        __syncthreads();
        __threadfence();
    }
}

// ---------------------------------------------------------------------------
// Host launcher
// ---------------------------------------------------------------------------
extern "C" void kernel_launch(void* const* d_in, const int* in_sizes, int n_in,
                              void* d_out, int out_size, void* d_ws, size_t ws_size,
                              hipStream_t stream) {
    (void)in_sizes; (void)n_in; (void)out_size;

    const float* x    = (const float*)d_in[0];
    const float* h0   = (const float*)d_in[1];   // [2, BB, HH]
    const float* wih0 = (const float*)d_in[2];
    const float* whh0 = (const float*)d_in[3];
    const float* bih0 = (const float*)d_in[4];
    const float* bhh0 = (const float*)d_in[5];
    const float* wih1 = (const float*)d_in[6];
    const float* whh1 = (const float*)d_in[7];
    const float* bih1 = (const float*)d_in[8];
    const float* bhh1 = (const float*)d_in[9];
    float* out = (float*)d_out;

    // Workspace carve-up (256B-aligned regions)
    char* ws = (char*)d_ws;
    size_t off = 0;
    auto carve = [&](size_t bytes) -> char* {
        char* p = ws + off;
        off += (bytes + 255) & ~(size_t)255;
        return p;
    };
    unsigned* ctr   = (unsigned*)carve(256);                       // 2 barrier counters
    __bf16*   xb    = (__bf16*)  carve((size_t)BB * TT * II * 2);  // x in bf16
    __bf16*   ob    = (__bf16*)  carve((size_t)BB * TT * HH * 2);  // layer-0 output bf16
    float*    pre   = (float*)   carve((size_t)BB * TT * HH * 4);  // pre-activations (reused)
    __bf16*   wih0b = (__bf16*)  carve((size_t)HH * II * 2);
    __bf16*   whh0b = (__bf16*)  carve((size_t)HH * HH * 2);
    __bf16*   wih1b = (__bf16*)  carve((size_t)HH * HH * 2);
    __bf16*   whh1b = (__bf16*)  carve((size_t)HH * HH * 2);
    __bf16*   h0p   = (__bf16*)  carve((size_t)BB * HH * 2);
    __bf16*   h0q   = (__bf16*)  carve((size_t)BB * HH * 2);
    __bf16*   h1p   = (__bf16*)  carve((size_t)BB * HH * 2);
    __bf16*   h1q   = (__bf16*)  carve((size_t)BB * HH * 2);
    if (off > ws_size) return;  // workspace too small; bail deterministically

    // 0) reset barrier counters (fresh every launch -> graph-replay safe)
    k_zero_u32<<<1, 64, 0, stream>>>(ctr, 2);

    // 1) f32 -> bf16 conversions
    {
        const int nx = BB * TT * II;
        k_cvt_bf16<<<(nx + 255) / 256, 256, 0, stream>>>(x, xb, nx);
        const int nh = BB * HH;
        k_cvt_bf16<<<(nh + 255) / 256, 256, 0, stream>>>(h0,      h0p, nh);
        k_cvt_bf16<<<(nh + 255) / 256, 256, 0, stream>>>(h0 + nh, h1p, nh);
        const int nw = HH * II;
        k_cvt_bf16<<<(nw + 255) / 256, 256, 0, stream>>>(wih0, wih0b, nw);
        k_cvt_bf16<<<(nw + 255) / 256, 256, 0, stream>>>(whh0, whh0b, nw);
        k_cvt_bf16<<<(nw + 255) / 256, 256, 0, stream>>>(wih1, wih1b, nw);
        k_cvt_bf16<<<(nw + 255) / 256, 256, 0, stream>>>(whh1, whh1b, nw);
    }

    const int gemm_blocks = ((BB * TT / 16) * (HH / 64)) / 8;  // 2048
    const size_t rnn_lds = (size_t)4 * 16 * CSTR * 2;          // 132,096 B

    // 2) layer 0: pre0 = x @ w_ih0^T + b_ih0 + b_hh0, then recurrence
    k_gemm_pre<<<gemm_blocks, 256, 0, stream>>>(xb, wih0b, bih0, bhh0, pre);
    k_rnn_layer<false><<<16, 256, rnn_lds, stream>>>(pre, whh0b, h0p, h0q, ob, nullptr, ctr + 0);

    // 3) layer 1: pre1 = out0 @ w_ih1^T + b_ih1 + b_hh1, then recurrence -> d_out
    k_gemm_pre<<<gemm_blocks, 256, 0, stream>>>(ob, wih1b, bih1, bhh1, pre);
    k_rnn_layer<true><<<16, 256, rnn_lds, stream>>>(pre, whh1b, h1p, h1q, nullptr, out, ctr + 1);
}
